// DecoderLayer_17360257810693
// MI455X (gfx1250) — compile-verified, matrix operand order checked by reading, not run on hardware
//
#include <hip/hip_runtime.h>
#include <hip/hip_bf16.h>

typedef __bf16 bf16_t;
typedef __attribute__((ext_vector_type(16))) __bf16 v16bf;
typedef __attribute__((ext_vector_type(8)))  __bf16 v8bf;
typedef __attribute__((ext_vector_type(8)))  float  v8f;

#define D_MODEL 1024
#define N_HEADS 16
#define DH      64
#define D_FF    4096
#define BB      4
#define LSEQ    1024
#define BL      (BB * LSEQ)
#define EPSF    1e-6f

static __device__ __forceinline__ v16bf cat8(v8bf lo, v8bf hi) {
    return __builtin_shufflevector(lo, hi, 0,1,2,3,4,5,6,7,8,9,10,11,12,13,14,15);
}
// Fragment load from a row-major LDS tile row: two 16B ds_load_b128.
static __device__ __forceinline__ v16bf ldfrag(const bf16_t* p) {
    return cat8(*(const v8bf*)p, *(const v8bf*)(p + 16));
}

// ---------------------------------------------------------------------------
// bf16 GEMM: C[M,N] = alpha * A[M,K] @ Bt[N,K]^T (+bias)(+relu)(+resid)
// Bt is pre-transposed (row n = B[:,n]). WG tile 64x256, 8 waves, wave tile
// 32x64 (2x4 WMMA 16x16x32 bf16). Ping-pong LDS double buffering: one barrier
// per K-step, next tile's global loads overlap current tile's WMMAs.
// OMODE: 0 = fp32 store, 1 = bf16 store, 2 = bf16 per-batch-transposed store.
// All dims are multiples of 64/256/32 in this workload -> no bounds checks.
// ---------------------------------------------------------------------------
template<bool BIAS, bool RELU, bool RESID, int OMODE>
__global__ __launch_bounds__(256) void gemm_bf16_kernel(
    const bf16_t* __restrict__ A, const bf16_t* __restrict__ Bt,
    int N, int K,
    const float* __restrict__ bias, const float* __restrict__ resid,
    float* __restrict__ outF, bf16_t* __restrict__ outB, float alpha)
{
    __shared__ __align__(16) bf16_t sA[2][64][40];
    __shared__ __align__(16) bf16_t sB[2][256][40];

    const int t    = threadIdx.x;
    const int wave = t >> 5, lane = t & 31;
    const int wm   = wave >> 2, wn = wave & 3;      // 2 x 4 wave grid
    const int lrow = lane & 15, lhz = lane >> 4;
    const int m0   = blockIdx.y * 64, n0 = blockIdx.x * 256;

    v8f acc[2][4] = {};

    const int ar = t >> 2, ac = (t & 3) * 8;        // A tile: 64 rows x 4 chunks
    const bf16_t* Ap = A  + (size_t)(m0 + ar) * K + ac;
    const bf16_t* Bp = Bt + (size_t)(n0 + t)  * K;  // B tile: 1 row (32 elems) per thread

    const int kc = 8 * lhz;
    auto compute = [&](int buf) {
        v16bf a0 = ldfrag(&sA[buf][wm * 32 + lrow][kc]);
        v16bf a1 = ldfrag(&sA[buf][wm * 32 + 16 + lrow][kc]);
#pragma unroll
        for (int j = 0; j < 4; ++j) {
            v16bf b = ldfrag(&sB[buf][wn * 64 + j * 16 + lrow][kc]);
            acc[0][j] = __builtin_amdgcn_wmma_f32_16x16x32_bf16(
                false, a0, false, b, (short)0, acc[0][j], false, false);
            acc[1][j] = __builtin_amdgcn_wmma_f32_16x16x32_bf16(
                false, a1, false, b, (short)0, acc[1][j], false, false);
        }
    };

    // prologue: stage tile 0
    {
        v8bf na  = *(const v8bf*)(Ap);
        v8bf nb0 = *(const v8bf*)(Bp);
        v8bf nb1 = *(const v8bf*)(Bp + 8);
        v8bf nb2 = *(const v8bf*)(Bp + 16);
        v8bf nb3 = *(const v8bf*)(Bp + 24);
        *(v8bf*)&sA[0][ar][ac] = na;
        *(v8bf*)&sB[0][t][0]   = nb0;
        *(v8bf*)&sB[0][t][8]   = nb1;
        *(v8bf*)&sB[0][t][16]  = nb2;
        *(v8bf*)&sB[0][t][24]  = nb3;
    }

    int cur = 0;
    for (int k0 = 32; k0 < K; k0 += 32) {
        __syncthreads();
        // next tile -> registers (in flight during WMMAs below)
        v8bf na  = *(const v8bf*)(Ap + k0);
        v8bf nb0 = *(const v8bf*)(Bp + k0);
        v8bf nb1 = *(const v8bf*)(Bp + k0 + 8);
        v8bf nb2 = *(const v8bf*)(Bp + k0 + 16);
        v8bf nb3 = *(const v8bf*)(Bp + k0 + 24);
        if (k0 + 32 < K) {  // pull tile k0+32 into GL2 (global_prefetch_b8)
            __builtin_prefetch(Ap + k0 + 32, 0, 1);
            __builtin_prefetch(Bp + k0 + 32, 0, 1);
        }
        compute(cur);
        int nxt = cur ^ 1;
        *(v8bf*)&sA[nxt][ar][ac] = na;
        *(v8bf*)&sB[nxt][t][0]   = nb0;
        *(v8bf*)&sB[nxt][t][8]   = nb1;
        *(v8bf*)&sB[nxt][t][16]  = nb2;
        *(v8bf*)&sB[nxt][t][24]  = nb3;
        cur = nxt;
    }
    __syncthreads();
    compute(cur);

    // ---- epilogue (fully branch-free: all flags are template params) ----
#pragma unroll
    for (int i = 0; i < 2; ++i)
#pragma unroll
        for (int j = 0; j < 4; ++j) {
            int n = n0 + wn * 64 + j * 16 + lrow;
            float bv = BIAS ? bias[n] : 0.0f;
#pragma unroll
            for (int v = 0; v < 8; ++v) {
                int m = m0 + wm * 32 + i * 16 + v + 8 * lhz;
                float c = acc[i][j][v] * alpha + bv;
                if (RELU) c = fmaxf(c, 0.0f);
                size_t idx = (size_t)m * N + n;
                if (RESID) c += resid[idx];
                if (OMODE == 0) {
                    outF[idx] = c;
                } else if (OMODE == 1) {
                    outB[idx] = (bf16_t)c;
                } else {  // per-batch transposed bf16 store: out[b][n][m%L]
                    int bi = m / LSEQ, ml = m % LSEQ;
                    outB[((size_t)bi * N + n) * LSEQ + ml] = (bf16_t)c;
                }
            }
        }
}

// ---------------------------------------------------------------------------
// Attention for one (batch, head, 16-query tile). Q,K row-major bf16 [B*L, D];
// Vt bf16 [B][D][Lk] (per-batch transposed V projection). Scores fp32 in a
// 64KB LDS tile; softmax fp32; ctx via WMMA with fp32->bf16 fragment cvt.
// Head 0 probabilities are written to `top` ([B][Lq][Lk] fp32).
// ---------------------------------------------------------------------------
template<int CAUSAL>
__global__ __launch_bounds__(256) void attention_kernel(
    const bf16_t* __restrict__ Q, const bf16_t* __restrict__ Kmat,
    const bf16_t* __restrict__ Vt, bf16_t* __restrict__ ctx,
    float* __restrict__ top, int Lq, int Lk)
{
    __shared__ __align__(16) float sc[16][1024];

    const int t    = threadIdx.x;
    const int wave = t >> 5, lane = t & 31;
    const int lrow = lane & 15, lhz = lane >> 4;
    const int q0   = blockIdx.x * 16;
    const int h    = blockIdx.y;
    const int bz   = blockIdx.z;

    // ---- phase A: scores = Q @ K^T (Q pre-scaled by 1/sqrt(dh)) ----
    v16bf aq[2];
    {
        const bf16_t* qp = Q + ((size_t)(bz * Lq + q0 + lrow)) * D_MODEL + h * DH;
#pragma unroll
        for (int ks = 0; ks < 2; ++ks) {
            int kb = ks * 32 + 8 * lhz;
            aq[ks] = cat8(*(const v8bf*)(qp + kb), *(const v8bf*)(qp + kb + 16));
        }
    }
    for (int nt = 0; nt < 8; ++nt) {
        int n0 = wave * 128 + nt * 16;
        const bf16_t* kp = Kmat + ((size_t)(bz * Lk + n0 + lrow)) * D_MODEL + h * DH;
        v8f acc = {};
#pragma unroll
        for (int ks = 0; ks < 2; ++ks) {
            int kb = ks * 32 + 8 * lhz;
            v16bf bk = cat8(*(const v8bf*)(kp + kb), *(const v8bf*)(kp + kb + 16));
            acc = __builtin_amdgcn_wmma_f32_16x16x32_bf16(
                false, aq[ks], false, bk, (short)0, acc, false, false);
        }
#pragma unroll
        for (int v = 0; v < 8; ++v)
            sc[v + 8 * lhz][n0 + lrow] = acc[v];
    }
    __syncthreads();

    // ---- phase B: masked softmax, 16 lanes per query row ----
    {
        int r = t >> 4, sub = t & 15;
        float* row = sc[r];
        int qa = q0 + r;
        float mx = -3.0e38f;
        for (int j = sub; j < Lk; j += 16) {
            float v = row[j];
            if (CAUSAL && j > qa) { v = -1.0e30f; row[j] = v; }
            mx = fmaxf(mx, v);
        }
#pragma unroll
        for (int o = 8; o > 0; o >>= 1) mx = fmaxf(mx, __shfl_xor(mx, o, 16));
        float sum = 0.0f;
        for (int j = sub; j < Lk; j += 16) {
            float e = __expf(row[j] - mx);
            row[j] = e;
            sum += e;
        }
#pragma unroll
        for (int o = 8; o > 0; o >>= 1) sum += __shfl_xor(sum, o, 16);
        float inv = 1.0f / sum;
        for (int j = sub; j < Lk; j += 16) {
            float p = row[j] * inv;
            row[j] = p;
            if (h == 0) top[((size_t)(bz * Lq + qa)) * Lk + j] = p;
        }
    }
    __syncthreads();

    // ---- phase C: ctx = P @ V; 4 waves cover the 4 dh/16 column tiles ----
    if (wave < 4) {
        const bf16_t* vp = Vt + ((size_t)bz * D_MODEL + h * DH + wave * 16 + lrow) * Lk;
        v8f acc = {};
        for (int ks = 0; ks < Lk / 32; ++ks) {
            int kb = ks * 32 + 8 * lhz;
            const float* pr = &sc[lrow][kb];
            v16bf ap;
#pragma unroll
            for (int e = 0; e < 8; ++e) ap[e] = (bf16_t)pr[e];
#pragma unroll
            for (int e = 0; e < 8; ++e) ap[8 + e] = (bf16_t)pr[16 + e];
            v16bf bv = cat8(*(const v8bf*)(vp + kb), *(const v8bf*)(vp + kb + 16));
            acc = __builtin_amdgcn_wmma_f32_16x16x32_bf16(
                false, ap, false, bv, (short)0, acc, false, false);
        }
#pragma unroll
        for (int v = 0; v < 8; ++v)
            ctx[((size_t)(bz * Lq + q0 + v + 8 * lhz)) * D_MODEL
                + h * DH + wave * 16 + lrow] = (bf16_t)acc[v];
    }
}

// ---------------------------------------------------------------------------
// Fp32 LayerNorm over D_MODEL, bf16 output. One 256-thread block per row.
// ---------------------------------------------------------------------------
__global__ __launch_bounds__(256) void layernorm_kernel(
    const float* __restrict__ x, const float* __restrict__ g,
    const float* __restrict__ b, bf16_t* __restrict__ out)
{
    __shared__ float red[16];
    const int row = blockIdx.x;
    const int t = threadIdx.x;
    const float* xp = x + (size_t)row * D_MODEL;
    float s = 0.f, s2 = 0.f;
    for (int j = t; j < D_MODEL; j += 256) { float v = xp[j]; s += v; s2 += v * v; }
#pragma unroll
    for (int o = 16; o > 0; o >>= 1) { s += __shfl_xor(s, o, 32); s2 += __shfl_xor(s2, o, 32); }
    if ((t & 31) == 0) { red[t >> 5] = s; red[8 + (t >> 5)] = s2; }
    __syncthreads();
    float S = 0.f, S2 = 0.f;
#pragma unroll
    for (int i = 0; i < 8; ++i) { S += red[i]; S2 += red[8 + i]; }
    float mu   = S / (float)D_MODEL;
    float var  = S2 / (float)D_MODEL - mu * mu;
    float rstd = rsqrtf(var + EPSF);
    bf16_t* op = out + (size_t)row * D_MODEL;
    for (int j = t; j < D_MODEL; j += 256)
        op[j] = (bf16_t)((xp[j] - mu) * rstd * g[j] + b[j]);
}

// ---------------------------------------------------------------------------
// fp32 W[Kd][Nd] -> bf16 Wt[Nd][Kd] (tiled transpose + convert)
// ---------------------------------------------------------------------------
__global__ void transpose_cvt_kernel(const float* __restrict__ W,
                                     bf16_t* __restrict__ Wt, int Kd, int Nd)
{
    __shared__ float tile[32][33];
    int n0 = blockIdx.x * 32, k0 = blockIdx.y * 32;
    for (int i = threadIdx.y; i < 32; i += 8)
        tile[i][threadIdx.x] = W[(size_t)(k0 + i) * Nd + n0 + threadIdx.x];
    __syncthreads();
    for (int i = threadIdx.y; i < 32; i += 8)
        Wt[(size_t)(n0 + i) * Kd + k0 + threadIdx.x] = (bf16_t)tile[threadIdx.x][i];
}

__global__ void cvt_bf16_kernel(const float* __restrict__ in,
                                bf16_t* __restrict__ out, int n)
{
    int i = blockIdx.x * blockDim.x + threadIdx.x;
    int stride = gridDim.x * blockDim.x;
    for (; i < n; i += stride) out[i] = (bf16_t)in[i];
}

// ---------------------------------------------------------------------------
extern "C" void kernel_launch(void* const* d_in, const int* in_sizes, int n_in,
                              void* d_out, int out_size, void* d_ws, size_t ws_size,
                              hipStream_t stream)
{
    (void)in_sizes; (void)n_in; (void)out_size; (void)ws_size;

    const float* dec    = (const float*)d_in[0];
    const float* enc    = (const float*)d_in[1];
    const float* sa_wq  = (const float*)d_in[4];
    const float* sa_wk  = (const float*)d_in[5];
    const float* sa_wv  = (const float*)d_in[6];
    const float* sa_wo  = (const float*)d_in[7];
    const float* ca_wq  = (const float*)d_in[8];
    const float* ca_wk  = (const float*)d_in[9];
    const float* ca_wv  = (const float*)d_in[10];
    const float* ca_wo  = (const float*)d_in[11];
    const float* ffn_w1 = (const float*)d_in[12];
    const float* ffn_b1 = (const float*)d_in[13];
    const float* ffn_w2 = (const float*)d_in[14];
    const float* ffn_b2 = (const float*)d_in[15];
    const float* ln1_g  = (const float*)d_in[16];
    const float* ln1_b  = (const float*)d_in[17];
    const float* ln2_g  = (const float*)d_in[18];
    const float* ln2_b  = (const float*)d_in[19];
    const float* lnf_g  = (const float*)d_in[20];
    const float* lnf_b  = (const float*)d_in[21];

    float* out_x  = (float*)d_out;
    float* out_sa = out_x  + (size_t)BL * D_MODEL;
    float* out_ca = out_sa + (size_t)BB * LSEQ * LSEQ;

    size_t off = 0;
    auto carve = [&](size_t bytes) -> void* {
        void* p = (char*)d_ws + off;
        off = (off + bytes + 255) & ~(size_t)255;
        return p;
    };
    const size_t DD = (size_t)D_MODEL * D_MODEL;
    bf16_t* t_sa_wq = (bf16_t*)carve(DD * 2);
    bf16_t* t_sa_wk = (bf16_t*)carve(DD * 2);
    bf16_t* t_sa_wv = (bf16_t*)carve(DD * 2);
    bf16_t* t_sa_wo = (bf16_t*)carve(DD * 2);
    bf16_t* t_ca_wq = (bf16_t*)carve(DD * 2);
    bf16_t* t_ca_wk = (bf16_t*)carve(DD * 2);
    bf16_t* t_ca_wv = (bf16_t*)carve(DD * 2);
    bf16_t* t_ca_wo = (bf16_t*)carve(DD * 2);
    bf16_t* t_w1    = (bf16_t*)carve((size_t)D_FF * D_MODEL * 2); // [FF][D]
    bf16_t* t_w2    = (bf16_t*)carve((size_t)D_MODEL * D_FF * 2); // [D][FF]
    bf16_t* enc_bf  = (bf16_t*)carve((size_t)BL * D_MODEL * 2);
    bf16_t* xn      = (bf16_t*)carve((size_t)BL * D_MODEL * 2);
    bf16_t* Qb      = (bf16_t*)carve((size_t)BL * D_MODEL * 2);
    bf16_t* Kb      = (bf16_t*)carve((size_t)BL * D_MODEL * 2);
    bf16_t* Vtb     = (bf16_t*)carve((size_t)BB * D_MODEL * LSEQ * 2);
    bf16_t* ctxb    = (bf16_t*)carve((size_t)BL * D_MODEL * 2);
    float*  x1      = (float*)carve((size_t)BL * D_MODEL * 4);
    float*  x2      = (float*)carve((size_t)BL * D_MODEL * 4);
    bf16_t* hbuf    = (bf16_t*)carve((size_t)BL * D_FF * 2);

    const dim3 tb(32, 8);
    const dim3 gDD(D_MODEL / 32, D_MODEL / 32);
    transpose_cvt_kernel<<<gDD, tb, 0, stream>>>(sa_wq, t_sa_wq, D_MODEL, D_MODEL);
    transpose_cvt_kernel<<<gDD, tb, 0, stream>>>(sa_wk, t_sa_wk, D_MODEL, D_MODEL);
    transpose_cvt_kernel<<<gDD, tb, 0, stream>>>(sa_wv, t_sa_wv, D_MODEL, D_MODEL);
    transpose_cvt_kernel<<<gDD, tb, 0, stream>>>(sa_wo, t_sa_wo, D_MODEL, D_MODEL);
    transpose_cvt_kernel<<<gDD, tb, 0, stream>>>(ca_wq, t_ca_wq, D_MODEL, D_MODEL);
    transpose_cvt_kernel<<<gDD, tb, 0, stream>>>(ca_wk, t_ca_wk, D_MODEL, D_MODEL);
    transpose_cvt_kernel<<<gDD, tb, 0, stream>>>(ca_wv, t_ca_wv, D_MODEL, D_MODEL);
    transpose_cvt_kernel<<<gDD, tb, 0, stream>>>(ca_wo, t_ca_wo, D_MODEL, D_MODEL);
    transpose_cvt_kernel<<<dim3(D_FF / 32, D_MODEL / 32), tb, 0, stream>>>(ffn_w1, t_w1, D_MODEL, D_FF);
    transpose_cvt_kernel<<<dim3(D_MODEL / 32, D_FF / 32), tb, 0, stream>>>(ffn_w2, t_w2, D_FF, D_MODEL);

    cvt_bf16_kernel<<<(BL * D_MODEL) / 1024, 256, 0, stream>>>(enc, enc_bf, BL * D_MODEL);

    const float qscale = 0.125f; // 1/sqrt(dh=64)
    const dim3 gemmD (D_MODEL / 256, BL / 64);
    const dim3 gemmFF(D_FF    / 256, BL / 64);
    const dim3 attnG (LSEQ / 16, N_HEADS, BB);

    // ---- self-attention ----
    layernorm_kernel<<<BL, 256, 0, stream>>>(dec, ln1_g, ln1_b, xn);
    gemm_bf16_kernel<false,false,false,1><<<gemmD, 256, 0, stream>>>(
        xn, t_sa_wq, D_MODEL, D_MODEL, nullptr, nullptr, nullptr, Qb, qscale);
    gemm_bf16_kernel<false,false,false,1><<<gemmD, 256, 0, stream>>>(
        xn, t_sa_wk, D_MODEL, D_MODEL, nullptr, nullptr, nullptr, Kb, 1.0f);
    gemm_bf16_kernel<false,false,false,2><<<gemmD, 256, 0, stream>>>(
        xn, t_sa_wv, D_MODEL, D_MODEL, nullptr, nullptr, nullptr, Vtb, 1.0f);
    attention_kernel<1><<<attnG, 256, 0, stream>>>(Qb, Kb, Vtb, ctxb, out_sa, LSEQ, LSEQ);
    gemm_bf16_kernel<false,false,true,0><<<gemmD, 256, 0, stream>>>(
        ctxb, t_sa_wo, D_MODEL, D_MODEL, nullptr, dec, x1, nullptr, 1.0f);

    // ---- cross-attention ----
    layernorm_kernel<<<BL, 256, 0, stream>>>(x1, ln2_g, ln2_b, xn);
    gemm_bf16_kernel<false,false,false,1><<<gemmD, 256, 0, stream>>>(
        xn, t_ca_wq, D_MODEL, D_MODEL, nullptr, nullptr, nullptr, Qb, qscale);
    gemm_bf16_kernel<false,false,false,1><<<gemmD, 256, 0, stream>>>(
        enc_bf, t_ca_wk, D_MODEL, D_MODEL, nullptr, nullptr, nullptr, Kb, 1.0f);
    gemm_bf16_kernel<false,false,false,2><<<gemmD, 256, 0, stream>>>(
        enc_bf, t_ca_wv, D_MODEL, D_MODEL, nullptr, nullptr, nullptr, Vtb, 1.0f);
    attention_kernel<0><<<attnG, 256, 0, stream>>>(Qb, Kb, Vtb, ctxb, out_ca, LSEQ, LSEQ);
    gemm_bf16_kernel<false,false,true,0><<<gemmD, 256, 0, stream>>>(
        ctxb, t_ca_wo, D_MODEL, D_MODEL, nullptr, x1, x2, nullptr, 1.0f);

    // ---- FFN (pre-LN inside, residual from x2) ----
    layernorm_kernel<<<BL, 256, 0, stream>>>(x2, lnf_g, lnf_b, xn);
    gemm_bf16_kernel<true,true,false,1><<<gemmFF, 256, 0, stream>>>(
        xn, t_w1, D_FF, D_MODEL, ffn_b1, nullptr, nullptr, hbuf, 1.0f);
    gemm_bf16_kernel<true,false,true,0><<<gemmD, 256, 0, stream>>>(
        hbuf, t_w2, D_MODEL, D_FF, ffn_b2, x2, out_x, nullptr, 1.0f);
}
